// GridBasedPooling_5488968204865
// MI455X (gfx1250) — matrix operand order repaired and checked by reading, not compiled, not run on hardware
//
#include <hip/hip_runtime.h>

typedef __attribute__((ext_vector_type(16))) __bf16 v16bf;
typedef __attribute__((ext_vector_type(8)))  __bf16 v8bf;
typedef __attribute__((ext_vector_type(8)))  float  v8f;

#define TRACKS 1024
#define HIDDEN 128
#define LATENT 16
#define CELLS  256            // 16*16 grid cells
#define KDIM   4096           // LATENT*CELLS (flattened grid = GEMM K)
#define OUTD   128
#define KB     (KDIM / 32)    // 128 k-blocks of 32 (bf16 WMMA K)
#define NB     (OUTD / 16)    // 8 column tiles -> 8 waves
#define ROWSTRIDE 4104        // KDIM + 8 bf16 pad: break 64-bank LDS conflicts

// K-permutation: the reference flatten order is k = lat*256 + cell, but a GEMM
// is invariant under any K permutation applied to BOTH A and B. We use
// k' = cell*16 + lat so each scattered enc[j] row is 32 CONTIGUOUS bytes in
// LDS (2x ds_store_b128 instead of 16x ds_store_b16). Bpack bakes the
// permutation into the B operand.

// Workspace layout:
//   [0, 1MB)      : Bpack  (KB * NB * 32 lanes * 16 bf16) — permuted W_emb, B-fragment layout
//   [1MB, +32KB)  : enc    (TRACKS * LATENT bf16)         — hidden_state @ W_enc + b_enc
#define BPACK_BYTES ((size_t)KB * NB * 32 * 16 * 2)

// ---------------------------------------------------------------------------
// Kernel 1: enc[i][l] = hidden[i,:] . W_enc[:,l] + b_enc[l]   (tiny, 4 MFLOP)
// ---------------------------------------------------------------------------
__global__ void enc_kernel(const float* __restrict__ hidden,
                           const float* __restrict__ Wenc,
                           const float* __restrict__ benc,
                           __bf16* __restrict__ enc) {
    int t = blockIdx.x * blockDim.x + threadIdx.x;   // 0 .. TRACKS*LATENT-1
    int i = t >> 4, l = t & 15;
    float acc = benc[l];
    const float* h = hidden + i * HIDDEN;
#pragma unroll 8
    for (int k = 0; k < HIDDEN; ++k) acc += h[k] * Wenc[k * LATENT + l];
    enc[t] = (__bf16)acc;
}

// ---------------------------------------------------------------------------
// Kernel 2: pack W_emb (f32, [K=4096][N=128] row-major) into bf16 B-fragment
// layout in contiguous K order (lanes 0-15 hold K' 0..15 of column N=lane, lanes
// 16-31 hold K' 16..31), applying the K permutation k' = cell*16 + lat,
// i.e. original k = (k'&15)*256 + (k'>>4). One thread = one lane fragment.
// ---------------------------------------------------------------------------
__global__ void pack_kernel(const float* __restrict__ Wemb,
                            __bf16* __restrict__ Bp) {
    int t = blockIdx.x * blockDim.x + threadIdx.x;   // 0 .. KB*NB*32-1
    int lane = t & 31;
    int nb   = (t >> 5) & (NB - 1);
    int kb   = t >> 8;
    int kbase = kb * 32 + ((lane >= 16) ? 16 : 0);
    int n     = nb * 16 + (lane & 15);
    v16bf v;
#pragma unroll
    for (int e = 0; e < 16; ++e) {
        int kp   = kbase + e;            // permuted K index
        int cell = kp >> 4;
        int lat  = kp & 15;
        v[e] = (__bf16)Wemb[(lat * CELLS + cell) * OUTD + n];
    }
    *(v16bf*)(Bp + (size_t)t * 16) = v;
}

// ---------------------------------------------------------------------------
// Kernel 3 (fused): per workgroup = 16 tracks.
//   phase 0: zero LDS occ tile, init winner table, load NaN-masked obs
//   phase 1: winner j per (track, cell)    (race == nondeterministic .set)
//   phase 2: gather enc[j] rows into the LDS occ tile (contiguous 32B/cell)
//   phase 3: 8 waves x v_wmma_f32_16x16x32_bf16 over K=4096, bias+relu store
// ---------------------------------------------------------------------------
__global__ void fused_kernel(const float* __restrict__ obs2,
                             const __bf16* __restrict__ enc,
                             const __bf16* __restrict__ Bp,
                             const float* __restrict__ bemb,
                             float* __restrict__ out) {
    extern __shared__ char smem[];
    __bf16* occ = (__bf16*)smem;                                   // 16*ROWSTRIDE bf16
    float*  obs = (float*)(smem + 16 * ROWSTRIDE * 2);             // TRACKS*2 f32
    short*  win = (short*)(smem + 16 * ROWSTRIDE * 2 + TRACKS * 8);// 16*CELLS i16

    const int tid = threadIdx.x;
    const int trkbase = blockIdx.x * 16;

    // ---- phase 0 -----------------------------------------------------------
    for (int w = tid; w < 16 * ROWSTRIDE / 2; w += 256)
        ((unsigned int*)occ)[w] = 0u;
    for (int w = tid; w < 16 * CELLS; w += 256)
        win[w] = -1;
    for (int j = tid; j < TRACKS; j += 256) {
        float x = obs2[2 * j], y = obs2[2 * j + 1];
        bool bad = (x != x) || (y != y);                 // isnan mask -> 0
        obs[2 * j]     = bad ? 0.f : x;
        obs[2 * j + 1] = bad ? 0.f : y;
    }
    __syncthreads();

    // ---- phase 1: winner index per cell ------------------------------------
    for (int idx = tid; idx < 16 * TRACKS; idx += 256) {
        int local = idx >> 10;
        int j     = idx & (TRACKS - 1);
        int i     = trkbase + local;
        if (j == i) continue;
        float ox = (obs[2 * j]     - obs[2 * i])     * 0.5f + 8.0f;
        float oy = (obs[2 * j + 1] - obs[2 * i + 1]) * 0.5f + 8.0f;
        if (ox >= 0.f && ox < 16.f && oy >= 0.f && oy < 16.f) {
            int cell = ((int)ox) * 16 + (int)oy;
            win[local * CELLS + cell] = (short)j;   // last writer wins (racy, like jax scatter)
        }
    }
    __syncthreads();

    // ---- phase 2: gather winners' encodings into LDS grid ------------------
    // Permuted K: row offset for a cell = cell*16 .. cell*16+15 (contiguous!)
    for (int idx = tid; idx < 16 * CELLS; idx += 256) {
        int local = idx >> 8;
        int cell  = idx & (CELLS - 1);
        int j     = win[idx];
        if (j >= 0) {
            const v8bf* ep = (const v8bf*)(enc + j * LATENT);
            v8bf* op = (v8bf*)(occ + local * ROWSTRIDE + cell * 16);
            op[0] = ep[0];                       // 2x ds_store_b128, conflict-free
            op[1] = ep[1];
        }
    }
    __syncthreads();

    // ---- phase 3: WMMA GEMM (16 x 4096) x (4096 x 128) ---------------------
    const int lane = tid & 31;
    const int nb   = tid >> 5;        // wave id -> output column tile
    const int half = lane >> 4;       // A layout: lane half selects K 16..31
    const int rowl = lane & 15;

    v8f c = {};
    const __bf16* bptr = Bp + ((size_t)nb * 32 + lane) * 16;
    const size_t  bstride = (size_t)NB * 32 * 16;    // elements per k-block
#pragma unroll 4
    for (int kb = 0; kb < KB; ++kb) {
        const __bf16* aptr = occ + rowl * ROWSTRIDE + kb * 32 + half * 16;
        v8bf alo = *(const v8bf*)(aptr);             // K' +0..7   (ds_load_b128)
        v8bf ahi = *(const v8bf*)(aptr + 8);         // K' +8..15  (ds_load_b128)
        v16bf a;
#pragma unroll
        for (int e = 0; e < 8; ++e) { a[e] = alo[e]; a[8 + e] = ahi[e]; }
        v16bf b = *(const v16bf*)(bptr);             // coalesced 32B/lane
        // Unconditional speculative prefetch 8 k-blocks (64KB) ahead:
        // OOB prefetches are silently dropped (ISA 10.5), and an
        // unrolled-loop-aligned unconditional form keeps the body branch-free.
        __builtin_prefetch(bptr + 8 * bstride, 0, 1);
        c = __builtin_amdgcn_wmma_f32_16x16x32_bf16(
                false, a, false, b, (short)0, c, false, false);
        bptr += bstride;
    }

    // C layout: VGPR v -> M = v + half*8, N = lane&15
    const int col  = nb * 16 + rowl;
    const float bias = bemb[col];
#pragma unroll
    for (int v = 0; v < 8; ++v) {
        int row = trkbase + v + half * 8;
        float r = c[v] + bias;
        out[row * OUTD + col] = r > 0.f ? r : 0.f;
    }
}

// ---------------------------------------------------------------------------
extern "C" void kernel_launch(void* const* d_in, const int* in_sizes, int n_in,
                              void* d_out, int out_size, void* d_ws, size_t ws_size,
                              hipStream_t stream) {
    const float* hidden = (const float*)d_in[0];
    // d_in[1] = obs1 (unused by the reference computation)
    const float* obs2   = (const float*)d_in[2];
    const float* Wenc   = (const float*)d_in[3];
    const float* benc   = (const float*)d_in[4];
    const float* Wemb   = (const float*)d_in[5];
    const float* bemb   = (const float*)d_in[6];
    float* out = (float*)d_out;

    __bf16* Bp  = (__bf16*)d_ws;
    __bf16* enc = (__bf16*)((char*)d_ws + BPACK_BYTES);

    pack_kernel<<<KB * NB * 32 / 256, 256, 0, stream>>>(Wemb, Bp);
    enc_kernel<<<TRACKS * LATENT / 256, 256, 0, stream>>>(hidden, Wenc, benc, enc);

    size_t lds = (size_t)16 * ROWSTRIDE * 2   // occ tile (bf16)
               + (size_t)TRACKS * 2 * 4       // masked obs
               + (size_t)16 * CELLS * 2;      // winner table
    fused_kernel<<<TRACKS / 16, 256, lds, stream>>>(obs2, enc, Bp, bemb, out);
}
